// filter_CNN_89610197664465
// MI455X (gfx1250) — matrix operand order repaired and checked by reading, not compiled, not run on hardware
//
#include <hip/hip_runtime.h>
#include <hip/hip_bf16.h>

// ---- problem dimensions (from reference) ----
#define Bn   128      // batch
#define Ln   2048     // sequence length == T
#define Cn   128      // conv out channels
#define Kn   7        // conv kernel
#define Hn   256      // hidden
#define NCn  10       // classes
#define Tn   2048
#define THR  1.0f

typedef __attribute__((ext_vector_type(16))) _Float16 v16h;
typedef __attribute__((ext_vector_type(8)))  _Float16 v8h;
typedef __attribute__((ext_vector_type(8)))  float    v8f;

// ---------------------------------------------------------------------------
// Prep kernel 1: Conv1d(in=1,out=128,k=7,pad=3) -> xc[t][b][c]  (f32)
// ---------------------------------------------------------------------------
__global__ __launch_bounds__(256) void conv1d_kernel(const float* __restrict__ x,
                                                     const float* __restrict__ w,
                                                     float* __restrict__ xc)
{
    int idx = blockIdx.x * 256 + threadIdx.x;      // t*B*C + b*C + c
    int c   = idx & (Cn - 1);
    int bt  = idx >> 7;
    int b   = bt & (Bn - 1);
    int t   = bt >> 7;
    float acc = 0.f;
#pragma unroll
    for (int k = 0; k < Kn; ++k) {
        int xi = t + k - 3;
        float xv = (xi >= 0 && xi < Ln) ? x[b * Ln + xi] : 0.f;
        acc += w[c * Kn + k] * xv;
    }
    xc[idx] = acc;
}

// ---------------------------------------------------------------------------
// Prep kernel 2: weights f32 -> f16; w2 padded 10 -> 16 rows (zeros)
// ---------------------------------------------------------------------------
__global__ __launch_bounds__(256) void wcvt_kernel(const float* __restrict__ w1,
                                                   const float* __restrict__ w2,
                                                   _Float16* __restrict__ w1h,
                                                   _Float16* __restrict__ w2h)
{
    int idx = blockIdx.x * 256 + threadIdx.x;
    if (idx < Hn * Cn) {
        w1h[idx] = (_Float16)w1[idx];
    } else {
        int j = idx - Hn * Cn;                      // 0 .. 16*256-1
        int n = j >> 8;
        int k = j & 255;
        w2h[j] = (n < NCn) ? (_Float16)w2[n * Hn + k] : (_Float16)0.f;
    }
}

// ---------------------------------------------------------------------------
// Main scan kernel: 8 blocks x 256 threads (8 waves of 32).
// Block handles a 16-batch WMMA tile for all 2048 time steps.
//   wave w owns N-tiles {2w, 2w+1} of H (layer 2); wave 0 also owns layer 3.
//   v1: registers (8 f32/thread); v2: WMMA accumulators (2 x v8f / wave);
//   v3 + spike-sum: WMMA accumulators in wave 0.
// ---------------------------------------------------------------------------
__global__ __launch_bounds__(256) void snn_scan_kernel(const float* __restrict__ xc,
                                                       const _Float16* __restrict__ w1h,
                                                       const _Float16* __restrict__ w2h,
                                                       float* __restrict__ out)
{
    __shared__ _Float16 s1buf[16 * Cn];   // spikes layer 1: [batch][C]   4 KB
    __shared__ _Float16 s2buf[16 * Hn];   // spikes layer 2: [batch][H]   8 KB

    const int tid  = threadIdx.x;
    const int lane = tid & 31;
    const int wv   = tid >> 5;
    const int blk  = blockIdx.x;          // 16-batch tile index, 0..7
    const int lm   = lane & 15;           // lane row within half-wave
    const int lh   = lane >> 4;           // half-wave select (ISA A/B layout)

    // ---- phase-A mapping: thread -> (batch row, 8 channels) ----
    const int pb = tid >> 4;              // 0..15
    const int pc = (tid & 15) << 3;       // channel base, 8 per thread
    float v1[8];
#pragma unroll
    for (int i = 0; i < 8; ++i) v1[i] = 0.f;

    // ---- preload layer-2 B fragments (w1^T) : resident whole scan ----
    // B[k][n] = w1[n][k]; lane holds column n, 16 consecutive k (32B load).
    v16h bfr[4][2];
#pragma unroll
    for (int kt = 0; kt < 4; ++kt)
#pragma unroll
        for (int j = 0; j < 2; ++j) {
            int n = (2 * wv + j) * 16 + lm;
            bfr[kt][j] = *(const v16h*)(w1h + n * Cn + kt * 32 + lh * 16);
        }

    // ---- layer-2 membrane state lives in the WMMA accumulators ----
    v8f v2acc[2];
#pragma unroll
    for (int j = 0; j < 2; ++j)
#pragma unroll
        for (int e = 0; e < 8; ++e) v2acc[j][e] = 0.f;

    // ---- wave 0: layer-3 B fragments (w2^T, padded to 16 cols) + state ----
    v16h cfr[8];
    v8f  v3acc, spksum;
#pragma unroll
    for (int e = 0; e < 8; ++e) { v3acc[e] = 0.f; spksum[e] = 0.f; }
    if (wv == 0) {
#pragma unroll
        for (int kt = 0; kt < 8; ++kt)
            cfr[kt] = *(const v16h*)(w2h + lm * Hn + kt * 32 + lh * 16);
    }

    const float* xrow0 = xc + (size_t)(blk * 16 + pb) * Cn + pc;

    for (int t = 0; t < Tn; ++t) {
        // ================= phase A: IF layer 1 =================
        const float* xr = xrow0 + (size_t)t * (Bn * Cn);
        float4 xa = ((const float4*)xr)[0];
        float4 xb = ((const float4*)xr)[1];
        float xv[8] = {xa.x, xa.y, xa.z, xa.w, xb.x, xb.y, xb.z, xb.w};
        __builtin_prefetch(xr + Bn * Cn, 0, 1);   // global_prefetch for t+1
        v8h sp;
#pragma unroll
        for (int i = 0; i < 8; ++i) {
            v1[i] += xv[i];
            bool s = (v1[i] - THR) >= 0.f;
            sp[i]  = s ? (_Float16)1.f : (_Float16)0.f;
            v1[i]  = s ? 0.f : v1[i];             // hard reset
        }
        *(v8h*)(s1buf + pb * Cn + pc) = sp;       // 16B ds store
        __syncthreads();

        // ================= phase B: s1 @ w1^T, IF layer 2 =================
        // A-fragment contiguous K order (matches B): lanes 0-15 K=0..15,
        // lanes 16-31 K=16..31 (two aligned 16B LDS loads each).
        v16h afr[4];
#pragma unroll
        for (int kt = 0; kt < 4; ++kt) {
            const _Float16* ap = s1buf + lm * Cn + kt * 32 + lh * 16;
            v8h lo = *(const v8h*)ap;
            v8h hi = *(const v8h*)(ap + 8);
#pragma unroll
            for (int i = 0; i < 8; ++i) { afr[kt][i] = lo[i]; afr[kt][8 + i] = hi[i]; }
        }
#pragma unroll
        for (int j = 0; j < 2; ++j) {
            v8f c = v2acc[j];
#pragma unroll
            for (int kt = 0; kt < 4; ++kt)
                c = __builtin_amdgcn_wmma_f32_16x16x32_f16(
                        false, afr[kt], false, bfr[kt][j],
                        (short)0, c, false, false);
            // spike + reset in the accumulator; D layout: VGPR e <-> M = e + 8*lh
            int hb = (2 * wv + j) * 16 + lm;
#pragma unroll
            for (int e = 0; e < 8; ++e) {
                float v = c[e];
                float s = ((v - THR) >= 0.f) ? 1.f : 0.f;
                s2buf[(e + lh * 8) * Hn + hb] = (_Float16)s;
                c[e] = (s != 0.f) ? 0.f : v;
            }
            v2acc[j] = c;
        }
        __syncthreads();

        // ================= phase C: s2 @ w2^T, IF layer 3 (wave 0) ========
        if (wv == 0) {
            v8f c3 = v3acc;
#pragma unroll
            for (int kt = 0; kt < 8; ++kt) {
                const _Float16* ap = s2buf + lm * Hn + kt * 32 + lh * 16;
                v8h lo = *(const v8h*)ap;
                v8h hi = *(const v8h*)(ap + 8);
                v16h a2;
#pragma unroll
                for (int i = 0; i < 8; ++i) { a2[i] = lo[i]; a2[8 + i] = hi[i]; }
                c3 = __builtin_amdgcn_wmma_f32_16x16x32_f16(
                         false, a2, false, cfr[kt],
                         (short)0, c3, false, false);
            }
#pragma unroll
            for (int e = 0; e < 8; ++e) {
                float v = c3[e];
                float s = ((v - THR) >= 0.f) ? 1.f : 0.f;
                spksum[e] += s;
                c3[e] = (s != 0.f) ? 0.f : v;
            }
            v3acc = c3;
        }
        __syncthreads();   // s1/s2 consumed -> safe to overwrite next step
    }

    // ---- epilogue: out[b][n] = spksum / T  (wave 0, D layout addressing) ----
    if (wv == 0 && lm < NCn) {
#pragma unroll
        for (int e = 0; e < 8; ++e) {
            int batch = blk * 16 + e + lh * 8;
            out[batch * NCn + lm] = spksum[e] * (1.f / (float)Tn);
        }
    }
}

// ---------------------------------------------------------------------------
// launcher
// ---------------------------------------------------------------------------
extern "C" void kernel_launch(void* const* d_in, const int* in_sizes, int n_in,
                              void* d_out, int out_size, void* d_ws, size_t ws_size,
                              hipStream_t stream)
{
    const float* x      = (const float*)d_in[0];   // [128,2048]
    const float* conv_w = (const float*)d_in[1];   // [128,1,7]
    const float* w1     = (const float*)d_in[2];   // [256,128]
    const float* w2     = (const float*)d_in[3];   // [10,256]
    float*       out    = (float*)d_out;           // [128,10]

    char* ws = (char*)d_ws;
    const size_t XC_BYTES  = (size_t)Tn * Bn * Cn * sizeof(float);   // 128 MiB
    const size_t W1H_BYTES = (size_t)Hn * Cn * sizeof(_Float16);     // 64 KiB
    float*    xc  = (float*)ws;
    _Float16* w1h = (_Float16*)(ws + XC_BYTES);
    _Float16* w2h = (_Float16*)(ws + XC_BYTES + W1H_BYTES);

    // conv: T*B*C = 33,554,432 elements
    conv1d_kernel<<<(Tn * Bn * Cn) / 256, 256, 0, stream>>>(x, conv_w, xc);
    // weight conversion: 32768 + 4096 = 36864 = 144*256
    wcvt_kernel<<<144, 256, 0, stream>>>(w1, w2, w1h, w2h);
    // sequential scan: 8 blocks of 16 batches, 8 waves each
    snn_scan_kernel<<<Bn / 16, 256, 0, stream>>>(xc, w1h, w2h, out);
}
